// Attention_42305427865835
// MI455X (gfx1250) — matrix operand order repaired and checked by reading, not compiled, not run on hardware
//
#include <hip/hip_runtime.h>
#include <hip/hip_bf16.h>

// CDNA5 / gfx1250 wave32 WMMA types
typedef __attribute__((ext_vector_type(16))) __bf16 v16bf;
typedef __attribute__((ext_vector_type(8)))  float  v8f;
typedef __attribute__((ext_vector_type(4)))  unsigned int v4u;
typedef __attribute__((ext_vector_type(8)))  int v8i;
typedef __attribute__((ext_vector_type(4)))  int v4i;

union FragBF { v16bf v; uint4 q[2]; };

#define B_   32
#define LIN_ 2048
#define EDIM 512
#define DDIM 512
#define NDF  1024           // N*D = d_flat width
#define ROWS 64             // encoder rows per workgroup

#if defined(__gfx1250__) && __has_builtin(__builtin_amdgcn_tensor_load_to_lds)
#define USE_TDM 1
#else
#define USE_TDM 0
#endif

__device__ __forceinline__ unsigned short f2bf(float f) {
    unsigned u = __float_as_uint(f);
    u += 0x7FFFu + ((u >> 16) & 1u);   // round-to-nearest-even
    return (unsigned short)(u >> 16);
}
__device__ __forceinline__ unsigned pk2(float x, float y) {
    return (unsigned)f2bf(x) | ((unsigned)f2bf(y) << 16);
}

// ---- Kernel 0: W1_e (rows 1024..1535 of W1) fp32 -> bf16, row-major [E][D] ----
__global__ __launch_bounds__(256) void convert_w_kernel(
    const float* __restrict__ W1, unsigned short* __restrict__ Wb) {
    int idx = blockIdx.x * 256 + threadIdx.x;            // 0 .. 512*512-1
    Wb[idx] = f2bf(W1[(size_t)NDF * DDIM + idx]);
}

// ---- Kernel 1: dec_proj[b][d] = d_flat[b] . W1_d[:,d] + b1[d] (fp32) ----
__global__ __launch_bounds__(256) void dec_proj_kernel(
    const float* __restrict__ dflat,   // [B, 1024]
    const float* __restrict__ W1,      // [1536, 512]
    const float* __restrict__ b1,      // [512]
    float* __restrict__ dec) {         // [B, 512]
    int idx = blockIdx.x * 256 + threadIdx.x;            // 0 .. 16383
    int b = idx >> 9;
    int d = idx & (DDIM - 1);
    float s = b1[d];
    const float* x = dflat + (size_t)b * NDF;
    #pragma unroll 8
    for (int i = 0; i < NDF; ++i)
        s = fmaf(x[i], W1[(size_t)i * DDIM + d], s);
    dec[idx] = s;
}

// ---- Kernel 2: fused enc@W1e + dec + tanh + (.w2) -> raw scores ----
__global__ __launch_bounds__(256) void fused_attn_kernel(
    const float* __restrict__ X,            // encoder_outputs [B*LIN, E] fp32
    const unsigned short* __restrict__ Wb,  // W1_e bf16 [E, D]
    const float* __restrict__ dec,          // [B, D]
    const float* __restrict__ w2,           // [D]
    float* __restrict__ scores) {           // [B*LIN] raw (pre-softmax)
    __shared__ float Xs[ROWS * EDIM];       // 128 KB fp32 encoder tile (TDM destination)
    __shared__ float dec_s[DDIM];
    __shared__ float w2_s[DDIM];
    __shared__ float score_s[ROWS];

    const int tid   = threadIdx.x;
    const int row0  = blockIdx.x * ROWS;    // 64-row blocks never straddle a batch (2048 % 64 == 0)
    const int batch = row0 >> 11;

    const int wave  = tid >> 5;
    const int lane  = tid & 31;

    if (tid < ROWS) score_s[tid] = 0.0f;
    for (int i = tid; i < DDIM; i += 256) {
        dec_s[i] = dec[batch * DDIM + i];
        w2_s[i]  = w2[i];
    }

#if USE_TDM
    // One TDM descriptor (ISA ch.8) DMAs the 64x512 fp32 tile HBM -> LDS,
    // tracked by TENSORcnt; issued by wave 0 only (EXEC-independent, per-wave op).
    if (wave == 0) {
        const unsigned long long ga = (unsigned long long)(X + (size_t)row0 * EDIM);
        const unsigned lds_addr = (unsigned)(size_t)&Xs[0];  // flat addr low 32 bits == LDS offset
        // D# group0: count=1 | lds_addr | global_addr | type=2 ("image")
        v4u g0 = { 1u,
                   lds_addr,
                   (unsigned)(ga & 0xFFFFFFFFull),
                   (unsigned)((ga >> 32) & 0x01FFFFFFull) | (2u << 30) };
        // D# group1: data_size=2 (4B) | tensor_dim0=512 | tensor_dim1=64 |
        //            tile_dim0=512 | tile_dim1=64 | tensor_dim0_stride=512
        v8i g1 = { (int)(2u << 16),          // workgroup_mask=0, data_size=2
                   (int)(512u << 16),        // tensor_dim0[15:0]=512 in bits 63:48
                   (int)(64u  << 16),        // tensor_dim0[31:16]=0, tensor_dim1[15:0]=64
                   (int)(512u << 16),        // tensor_dim1[31:16]=0, tile_dim0=512
                   (int)64,                  // tile_dim1=64, tile_dim2=0
                   (int)512,                 // tensor_dim0_stride[31:0]=512
                   0,                        // stride0[47:32]=0, tensor_dim1_stride lo=0
                   0 };
        v4i g2 = { 0, 0, 0, 0 };             // 2D tensor: groups 2/3 unused
        v4i g3 = { 0, 0, 0, 0 };
#if __clang_major__ >= 23
        v8i g4 = { 0, 0, 0, 0, 0, 0, 0, 0 };
        __builtin_amdgcn_tensor_load_to_lds(g0, g1, g2, g3, g4, 0);
#else
        __builtin_amdgcn_tensor_load_to_lds(g0, g1, g2, g3, 0);
#endif
        __builtin_amdgcn_s_wait_tensorcnt(0);
    }
#else
    // Fallback: cooperative float4 staging
    {
        const float4* __restrict__ Xv = (const float4*)(X + (size_t)row0 * EDIM);
        float4* Xs4 = (float4*)Xs;
        for (int i = tid; i < ROWS * EDIM / 4; i += 256) Xs4[i] = Xv[i];
    }
#endif
    __syncthreads();

    const int rt    = wave & 3;     // row tile 0..3 (16 rows each)
    const int nhalf = wave >> 2;    // N half: 0 => d[0..255], 1 => d[256..511]
    const int m     = lane & 15;    // M index within tile (A rows / C columns)
    const int hi    = lane >> 4;    // lane-half selector per ISA layouts

    float score_acc[8];
    #pragma unroll
    for (int v = 0; v < 8; ++v) score_acc[v] = 0.0f;

    for (int ntile = 0; ntile < 16; ++ntile) {
        const int n0 = (nhalf * 16 + ntile) * 16;
        if (ntile < 15) {  // L2-side prefetch of next B tile -> global_prefetch_b8
            const int n1 = n0 + 16;
            __builtin_prefetch(&Wb[(size_t)(m + hi * 16) * DDIM + n1], 0, 3);
        }
        v8f acc = {};
        #pragma unroll
        for (int ks = 0; ks < 16; ++ks) {
            const int k0 = ks * 32;
            FragBF a, b;
            // A 16x32 bf16 (ISA 7.12.2): lane m (hi=0) holds K {k0..k0+7, k0+16..k0+23};
            // lane m (hi=1) holds K {k0+8..k0+15, k0+24..k0+31}.  Convert f32 LDS -> bf16
            // at build time; loop-invariant over ntile so the compiler hoists these.
            const float* ap = &Xs[(rt * 16 + m) * EDIM + k0 + hi * 8];
            float4 f0 = *(const float4*)(ap);
            float4 f1 = *(const float4*)(ap + 4);
            float4 f2 = *(const float4*)(ap + 16);
            float4 f3 = *(const float4*)(ap + 20);
            a.q[0].x = pk2(f0.x, f0.y); a.q[0].y = pk2(f0.z, f0.w);
            a.q[0].z = pk2(f1.x, f1.y); a.q[0].w = pk2(f1.z, f1.w);
            a.q[1].x = pk2(f2.x, f2.y); a.q[1].y = pk2(f2.z, f2.w);
            a.q[1].z = pk2(f3.x, f3.y); a.q[1].w = pk2(f3.z, f3.w);
            // B 32x16 bf16: lane column N = n0 + (lane&15); K order matches A (k0+8*hi+{0..7}, k0+16+8*hi+{0..7}).
            const unsigned short* bp = &Wb[(size_t)(k0 + hi * 8) * DDIM + n0 + m];
            unsigned* bw = (unsigned*)&b.q[0];
            #pragma unroll
            for (int j = 0; j < 8; ++j) {
                const int kk = (j < 4) ? 2 * j : 16 + 2 * (j - 4);
                bw[j] = (unsigned)bp[(size_t)kk * DDIM] | ((unsigned)bp[(size_t)(kk + 1) * DDIM] << 16);
            }
            acc = __builtin_amdgcn_wmma_f32_16x16x32_bf16(
                      false, a.v, false, b.v, (short)0, acc, false, false);
        }
        // Epilogue: C layout => column N = n0 + (lane&15), row M = v + 8*hi.
        const float dv = dec_s[n0 + m];
        const float wv = w2_s[n0 + m];
        #pragma unroll
        for (int v = 0; v < 8; ++v) {
            float t = tanhf(acc[v] + dv) * wv;
            // reduce over the 16 N-lanes of this half-wave
            t += __shfl_xor(t, 1, 32);
            t += __shfl_xor(t, 2, 32);
            t += __shfl_xor(t, 4, 32);
            t += __shfl_xor(t, 8, 32);
            score_acc[v] += t;
        }
    }
    // lanes 0 and 16 hold the per-row partial sums for M=v and M=v+8
    if (m == 0) {
        #pragma unroll
        for (int v = 0; v < 8; ++v)
            atomicAdd(&score_s[rt * 16 + v + 8 * hi], score_acc[v]);
    }
    __syncthreads();
    if (tid < ROWS) scores[row0 + tid] = score_s[tid];
}

// ---- Kernel 3: softmax over LIN=2048 per batch, in place on d_out ----
__global__ __launch_bounds__(256) void softmax_kernel(float* __restrict__ out) {
    __shared__ float red[256];
    const int b = blockIdx.x;
    float* s = out + (size_t)b * LIN_;
    float vals[8];
    float mx = -INFINITY;
    #pragma unroll
    for (int i = 0; i < 8; ++i) {
        vals[i] = s[threadIdx.x + i * 256];
        mx = fmaxf(mx, vals[i]);
    }
    red[threadIdx.x] = mx;
    __syncthreads();
    for (int off = 128; off > 0; off >>= 1) {
        if (threadIdx.x < off)
            red[threadIdx.x] = fmaxf(red[threadIdx.x], red[threadIdx.x + off]);
        __syncthreads();
    }
    mx = red[0];
    __syncthreads();
    float sum = 0.0f;
    #pragma unroll
    for (int i = 0; i < 8; ++i) {
        vals[i] = __expf(vals[i] - mx);
        sum += vals[i];
    }
    red[threadIdx.x] = sum;
    __syncthreads();
    for (int off = 128; off > 0; off >>= 1) {
        if (threadIdx.x < off)
            red[threadIdx.x] += red[threadIdx.x + off];
        __syncthreads();
    }
    const float inv = 1.0f / red[0];
    #pragma unroll
    for (int i = 0; i < 8; ++i)
        s[threadIdx.x + i * 256] = vals[i] * inv;
}

extern "C" void kernel_launch(void* const* d_in, const int* in_sizes, int n_in,
                              void* d_out, int out_size, void* d_ws, size_t ws_size,
                              hipStream_t stream) {
    const float* d_hidden = (const float*)d_in[0];  // (32,2,512) == d_flat (32,1024)
    const float* enc      = (const float*)d_in[1];  // (32,2048,512)
    const float* W1       = (const float*)d_in[2];  // (1536,512)
    const float* b1       = (const float*)d_in[3];  // (512)
    const float* w2       = (const float*)d_in[4];  // (512)
    float* out = (float*)d_out;                     // (32,2048) fp32

    // workspace layout: [Wb bf16 512x512 = 524288 B][dec fp32 32x512 = 65536 B]
    unsigned short* Wb = (unsigned short*)d_ws;
    float* dec = (float*)((char*)d_ws + (size_t)EDIM * DDIM * sizeof(unsigned short));

    convert_w_kernel<<<(EDIM * DDIM) / 256, 256, 0, stream>>>(W1, Wb);
    dec_proj_kernel<<<(B_ * DDIM) / 256, 256, 0, stream>>>(d_hidden, W1, b1, dec);
    fused_attn_kernel<<<(B_ * LIN_) / ROWS, 256, 0, stream>>>(enc, Wb, dec, w2, out);
    softmax_kernel<<<B_, 256, 0, stream>>>(out);
}
